// GridSelfAttention_53197464929136
// MI455X (gfx1250) — matrix operand order repaired and checked by reading, not compiled, not run on hardware
//
#include <hip/hip_runtime.h>

#define NTOK 384
#define CPAIR 128
#define NHEAD 4
#define HDIM 32
#define NN (NTOK * NTOK) /* 147456 */

typedef __attribute__((ext_vector_type(16))) _Float16 v16h;
typedef __attribute__((ext_vector_type(8)))  _Float16 v8h;
typedef __attribute__((ext_vector_type(8)))  float    v8f;

union H16 { v16h v; v8h h[2]; };

// ---------------- workspace layout (in _Float16 units) ----------------
static constexpr size_t OFF_WQ    = 0;                       // 128x128
static constexpr size_t OFF_WK    = OFF_WQ    + 16384;
static constexpr size_t OFF_WV    = OFF_WK    + 16384;
static constexpr size_t OFF_WGT   = OFF_WV    + 16384;       // gate weight
static constexpr size_t OFF_WO    = OFF_WGT   + 16384;
static constexpr size_t OFF_WB    = OFF_WO    + 16384;       // bias w, padded 16x128
static constexpr size_t OFF_Q     = OFF_WB    + 2048;        // [t][128] f16
static constexpr size_t OFF_K     = OFF_Q     + (size_t)NN * 128;
static constexpr size_t OFF_VT    = OFF_K     + (size_t)NN * 128;   // [b][h][d][key]
static constexpr size_t OFF_GATE  = OFF_VT    + (size_t)NN * 128;   // sigmoid applied
static constexpr size_t OFF_WGOUT = OFF_GATE  + (size_t)NN * 128;   // wa*gate
static constexpr size_t HALVES_TOTAL = OFF_WGOUT + (size_t)NN * 128;
// f32 BIAS [h][q][k] lives after HALVES_TOTAL halves.

// ---------------- WMMA helpers (gfx1250 wave32) ----------------
static __device__ inline v8f wmma32(v16h a, v16h b, v8f c) {
  // v_wmma_f32_16x16x32_f16 : D = A(16x32) x B(32x16) + C
  return __builtin_amdgcn_wmma_f32_16x16x32_f16(false, a, false, b, (short)0, c,
                                                false, false);
}

// A operand: lane holds row (lane&15); halves {c0+hi8..+7, c0+16+hi8..+7}
static __device__ inline v16h frag_a(const _Float16* base, int stride, int lane, int c0) {
  const int r  = lane & 15;
  const int hi = (lane & 16) ? 8 : 0;
  const _Float16* p0 = base + (size_t)r * stride + c0 + hi;
  H16 u;
  u.h[0] = *(const v8h*)p0;
  u.h[1] = *(const v8h*)(p0 + 16);
  return u.v;
}

// B operand: lane holds column (lane&15); halves {c0+hi8..+7, c0+16+hi8..+7}
static __device__ inline v16h frag_b(const _Float16* base, int stride, int lane, int c0) {
  const int n  = lane & 15;
  const int hi = (lane & 16) ? 8 : 0;
  const _Float16* p = base + (size_t)n * stride + c0 + hi;
  H16 u;
  u.h[0] = *(const v8h*)p;
  u.h[1] = *(const v8h*)(p + 16);
  return u.v;
}

// A operand built from f32 LDS with on-the-fly f32->f16 convert
static __device__ inline v16h frag_a_f32(const float* base, int stride, int lane, int c0) {
  const int r  = lane & 15;
  const int hi = (lane & 16) ? 8 : 0;
  const float* p0 = base + r * stride + c0 + hi;
  const float* p1 = p0 + 16;
  v16h a;
#pragma unroll
  for (int i = 0; i < 8; ++i) { a[i] = (_Float16)p0[i]; a[i + 8] = (_Float16)p1[i]; }
  return a;
}

// ---------------- kernel 0: weight conversion ----------------
__global__ void prep_weights(const float* wq, const float* wk, const float* wv,
                             const float* wbias, const float* wgate, const float* wout,
                             _Float16* H) {
  const float qs = 0.17677669529663687f; // 1/sqrt(HDIM)
  const int total = 5 * 16384 + 2048;
  for (int i = blockIdx.x * blockDim.x + threadIdx.x; i < total;
       i += gridDim.x * blockDim.x) {
    if (i < 16384)            H[OFF_WQ  + i]             = (_Float16)(wq[i] * qs);
    else if (i < 2 * 16384)   H[OFF_WK  + i - 16384]     = (_Float16)wk[i - 16384];
    else if (i < 3 * 16384)   H[OFF_WV  + i - 2 * 16384] = (_Float16)wv[i - 2 * 16384];
    else if (i < 4 * 16384)   H[OFF_WGT + i - 3 * 16384] = (_Float16)wgate[i - 3 * 16384];
    else if (i < 5 * 16384)   H[OFF_WO  + i - 4 * 16384] = (_Float16)wout[i - 4 * 16384];
    else {
      int j = i - 5 * 16384;           // padded [16][128]
      int o = j >> 7;
      H[OFF_WB + j] = (o < 4) ? (_Float16)wbias[j] : (_Float16)0.0f;
    }
  }
}

// ---------------- kernel 1: LayerNorm + q/k/v/gate/bias projections --------
__global__ __launch_bounds__(256) void ln_proj(const float* __restrict__ pair,
                                               const float* __restrict__ lnw,
                                               const float* __restrict__ lnb,
                                               _Float16* H, float* BIAS) {
  __shared__ _Float16 xs[64 * 136];     // 64 tokens x 128 ch (stride 136)
  __shared__ float rsum[64][4];
  __shared__ float rsq[64][4];

  const int tid = threadIdx.x;
  const size_t t0 = (size_t)blockIdx.x * 64;
  const int tk = tid >> 2;              // token in block
  const int qd = tid & 3;               // channel quarter

  // ---- LayerNorm (4 threads per token) ----
  float vals[32];
  const float* psrc = pair + (t0 + tk) * 128 + qd * 32;
  // prefetch next block's slice of pair into caches (gfx1250 global_prefetch)
  __builtin_prefetch(psrc + 64 * 128, 0, 1);
  float s = 0.f, sq = 0.f;
#pragma unroll
  for (int i = 0; i < 8; ++i) {
    float4 f = ((const float4*)psrc)[i];
    vals[i * 4 + 0] = f.x; vals[i * 4 + 1] = f.y;
    vals[i * 4 + 2] = f.z; vals[i * 4 + 3] = f.w;
    s  += f.x + f.y + f.z + f.w;
    sq += f.x * f.x + f.y * f.y + f.z * f.z + f.w * f.w;
  }
  rsum[tk][qd] = s; rsq[tk][qd] = sq;
  __syncthreads();
  const float S  = rsum[tk][0] + rsum[tk][1] + rsum[tk][2] + rsum[tk][3];
  const float SQ = rsq[tk][0] + rsq[tk][1] + rsq[tk][2] + rsq[tk][3];
  const float mean = S * (1.f / 128.f);
  const float var  = SQ * (1.f / 128.f) - mean * mean;
  const float rstd = rsqrtf(var + 1e-5f);
#pragma unroll
  for (int i = 0; i < 32; ++i) {
    const int c = qd * 32 + i;
    xs[tk * 136 + c] = (_Float16)((vals[i] - mean) * rstd * lnw[c] + lnb[c]);
  }
  __syncthreads();

  // ---- projections: 8 waves, wave w -> row-tile w&3, col-tiles (w>>2)+2k ----
  // readfirstlane: wave id is wave-uniform -> force SGPR so the per-kind
  // branches below become scalar (s_cbranch) instead of exec-mask dances.
  const int wave = __builtin_amdgcn_readfirstlane(tid >> 5);
  const int lane = tid & 31;
  const int rt = wave & 3;
  const int n = lane & 15;
  const int hi8 = (lane & 16) ? 8 : 0;
  const int b = (int)(t0 / 384);
  const int keybase = (int)(t0 % 384) + rt * 16 + hi8;
  const int trow0 = (int)t0 + rt * 16;

  v16h A[4];
#pragma unroll
  for (int c = 0; c < 4; ++c) A[c] = frag_a(xs + rt * 16 * 136, 136, lane, c * 32);

  for (int ct = (wave >> 2); ct < 33; ct += 2) {
    const _Float16* wmat; int kind; int o0;
    if (ct < 8)       { wmat = H + OFF_WQ;  kind = 0; o0 = ct * 16; }
    else if (ct < 16) { wmat = H + OFF_WK;  kind = 1; o0 = (ct - 8) * 16; }
    else if (ct < 24) { wmat = H + OFF_WV;  kind = 2; o0 = (ct - 16) * 16; }
    else if (ct < 32) { wmat = H + OFF_WGT; kind = 3; o0 = (ct - 24) * 16; }
    else              { wmat = H + OFF_WB;  kind = 4; o0 = 0; }

    // materialize all 4 B fragments first -> loads issue as one clause,
    // then the 4 WMMAs run back-to-back
    v16h Bf[4];
#pragma unroll
    for (int c = 0; c < 4; ++c)
      Bf[c] = frag_b(wmat + (size_t)o0 * 128, 128, lane, c * 32);
    v8f acc = {};
#pragma unroll
    for (int c = 0; c < 4; ++c) acc = wmma32(A[c], Bf[c], acc);

    if (kind == 0 || kind == 1 || kind == 3) {            // Q / K / GATE, [t][128]
      _Float16* dst = H + (kind == 0 ? OFF_Q : kind == 1 ? OFF_K : OFF_GATE);
#pragma unroll
      for (int j = 0; j < 8; ++j) {
        const int t = trow0 + j + hi8;
        float vv = acc[j];
        if (kind == 3) vv = 1.f / (1.f + __expf(-vv));    // sigmoid gate
        dst[(size_t)t * 128 + o0 + n] = (_Float16)vv;
      }
    } else if (kind == 2) {                               // V transposed: [b][h][d][key]
      const int o = o0 + n, hh = o >> 5, d = o & 31;
      _Float16* dst = H + OFF_VT + ((size_t)(b * 4 + hh) * 32 + d) * 384 + keybase;
#pragma unroll
      for (int j = 0; j < 8; j += 2) {
        union { _Float16 h[2]; unsigned u; } pk;
        pk.h[0] = (_Float16)acc[j]; pk.h[1] = (_Float16)acc[j + 1];
        *(unsigned*)(dst + j) = pk.u;
      }
    } else {                                              // BIAS f32, [h][q][k]
      if (n < 4) {
#pragma unroll
        for (int j = 0; j < 8; ++j)
          BIAS[(size_t)n * NN + (trow0 + j + hi8)] = acc[j];
      }
    }
  }
}

// ---------------- kernel 2: attention, one wave per (b, h, 16-query tile) --
// grid = (24 q-tiles, 4 heads, 384 rows)
__global__ __launch_bounds__(32) void attn(_Float16* H, const float* __restrict__ BIAS) {
  __shared__ float ss[16 * 388];        // 16 rows x 384 keys (stride 388)

  const int qt = blockIdx.x;
  const int h  = blockIdx.y;
  const int b  = blockIdx.z;
  const int lane = threadIdx.x;
  const int n = lane & 15;
  const int hi8 = (lane & 16) ? 8 : 0;
  const int qpos0 = qt * 16;

  const _Float16* Qb   = H + OFF_Q;
  const _Float16* Kb   = H + OFF_K;
  const _Float16* VT   = H + OFF_VT;
  const _Float16* GATE = H + OFF_GATE;
  _Float16*       WG   = H + OFF_WGOUT;

  // q fragment reused for every key tile (K = 32 = full head dim)
  const v16h qf = frag_a(Qb + ((size_t)(b * 384 + qpos0) * 128 + h * 32), 128, lane, 0);

  // ---- QK^T + bias, software-pipelined over key tiles ----
  const _Float16* kbase = Kb + ((size_t)(b * 384) * 128 + h * 32);
  v16h kf = frag_b(kbase, 128, lane, 0);
  for (int kt = 0; kt < 24; ++kt) {
    const v16h kcur = kf;
    const int ktn = (kt < 23) ? kt + 1 : 23;
    kf = frag_b(kbase + (size_t)ktn * 16 * 128, 128, lane, 0);
    v8f sc = {};
    sc = wmma32(qf, kcur, sc);
    const float* bp = BIAS + (size_t)h * NN + (size_t)qpos0 * 384 + kt * 16 + n;
#pragma unroll
    for (int j = 0; j < 8; ++j) {
      const int m = j + hi8;
      ss[m * 388 + kt * 16 + n] = sc[j] + bp[(size_t)m * 384];
    }
  }
  __syncthreads();

  // ---- softmax: lane covers row (lane&15), half (lane>>4) of 384 cols ----
  const int r = lane & 15;
  const int c0 = (lane & 16) ? 192 : 0;
  float* roww = ss + r * 388 + c0;
  float mx = -3.0e38f;
#pragma unroll 4
  for (int i = 0; i < 48; ++i) {
    float4 f = ((const float4*)roww)[i];
    mx = fmaxf(mx, fmaxf(fmaxf(f.x, f.y), fmaxf(f.z, f.w)));
  }
  mx = fmaxf(mx, __shfl_xor(mx, 16, 32));
  float sum = 0.f;
#pragma unroll 4
  for (int i = 0; i < 48; ++i) {
    float4 f = ((const float4*)roww)[i];
    f.x = __expf(f.x - mx); f.y = __expf(f.y - mx);
    f.z = __expf(f.z - mx); f.w = __expf(f.w - mx);
    sum += f.x + f.y + f.z + f.w;
    ((float4*)roww)[i] = f;
  }
  sum += __shfl_xor(sum, 16, 32);
  const float inv = 1.f / sum;
  __syncthreads();

  // ---- PV: P (from LDS, cvt to f16) x V^T, software-pipelined V loads ----
  const _Float16* vbase = VT + (size_t)((b * 4 + h) * 32) * 384;
#pragma unroll
  for (int dt = 0; dt < 2; ++dt) {
    const _Float16* vx = vbase + (size_t)dt * 16 * 384;
    v16h vb = frag_b(vx, 384, lane, 0);
    v8f acc = {};
#pragma unroll
    for (int kc = 0; kc < 12; ++kc) {
      const v16h vcur = vb;
      const int kn = (kc < 11) ? kc + 1 : 11;
      vb = frag_b(vx, 384, lane, kn * 32);
      const v16h pa = frag_a_f32(ss, 388, lane, kc * 32);
      acc = wmma32(pa, vcur, acc);
    }
#pragma unroll
    for (int j = 0; j < 8; ++j) {
      const int m = j + hi8;
      const float invm = __shfl(inv, m, 32);
      const size_t t = (size_t)(b * 384 + qpos0 + m);
      const int o = h * 32 + dt * 16 + n;
      const float g = (float)GATE[t * 128 + o];
      WG[t * 128 + o] = (_Float16)(acc[j] * invm * g);
    }
  }
}

// ---------------- kernel 3: output projection ----------------
__global__ __launch_bounds__(256) void out_proj(const _Float16* __restrict__ H,
                                                float* __restrict__ out) {
  __shared__ _Float16 wgs[64 * 136];

  const int tid = threadIdx.x;
  const size_t t0 = (size_t)blockIdx.x * 64;
  const int tk = tid >> 2, qd = tid & 3;
  const _Float16* src = H + OFF_WGOUT + (t0 + tk) * 128 + qd * 32;
#pragma unroll
  for (int i = 0; i < 4; ++i)
    *(v8h*)(wgs + tk * 136 + qd * 32 + i * 8) = *(const v8h*)(src + i * 8);
  __syncthreads();

  const int wave = __builtin_amdgcn_readfirstlane(tid >> 5);
  const int lane = tid & 31;
  const int n = lane & 15, hi8 = (lane & 16) ? 8 : 0;
  const int ct = wave;                               // 8 column tiles
  const _Float16* wo = H + OFF_WO + (size_t)ct * 16 * 128;

  v16h B[4];
#pragma unroll
  for (int c = 0; c < 4; ++c) B[c] = frag_b(wo, 128, lane, c * 32);

#pragma unroll
  for (int rt = 0; rt < 4; ++rt) {
    v8f acc = {};
#pragma unroll
    for (int c = 0; c < 4; ++c)
      acc = wmma32(frag_a(wgs + rt * 16 * 136, 136, lane, c * 32), B[c], acc);
#pragma unroll
    for (int j = 0; j < 8; ++j) {
      const size_t t = t0 + rt * 16 + j + hi8;
      out[t * 128 + ct * 16 + n] = acc[j];
    }
  }
}

// ---------------- host launch ----------------
extern "C" void kernel_launch(void* const* d_in, const int* in_sizes, int n_in,
                              void* d_out, int out_size, void* d_ws, size_t ws_size,
                              hipStream_t stream) {
  const float* pair  = (const float*)d_in[0];
  const float* lnw   = (const float*)d_in[1];
  const float* lnb   = (const float*)d_in[2];
  const float* wq    = (const float*)d_in[3];
  const float* wk    = (const float*)d_in[4];
  const float* wv    = (const float*)d_in[5];
  const float* wbias = (const float*)d_in[6];
  const float* wgate = (const float*)d_in[7];
  const float* wout  = (const float*)d_in[8];

  _Float16* H = (_Float16*)d_ws;
  float* BIAS = (float*)(H + HALVES_TOTAL);

  prep_weights<<<128, 256, 0, stream>>>(wq, wk, wv, wbias, wgate, wout, H);
  ln_proj<<<NN / 64, 256, 0, stream>>>(pair, lnw, lnb, H, BIAS);
  attn<<<dim3(24, NHEAD, NTOK), 32, 0, stream>>>(H, BIAS);
  out_proj<<<NN / 64, 256, 0, stream>>>(H, (float*)d_out);
}
